// DeformConv2d_33595234189730
// MI455X (gfx1250) — matrix operand order repaired and checked by reading, not compiled, not run on hardware
//
#include <hip/hip_runtime.h>
#include <hip/hip_bf16.h>

// ---------------------------------------------------------------------------
// Deformable Conv2d for MI455X (gfx1250, wave32).
//   Stage 1: offset conv (VALU, ~1.4 GFLOP).
//   Stage 2: weight permute + bf16 convert -> ws  (K order k = kk*256 + c).
//   Stage 3: fused bilinear-im2col + GEMM, v_wmma_f32_16x16x32_bf16,
//            double-buffered LDS tiles, async global->LDS weight staging.
// ---------------------------------------------------------------------------

typedef __bf16 v16bf __attribute__((ext_vector_type(16)));
typedef __bf16 v8bf  __attribute__((ext_vector_type(8)));
typedef float  v8f   __attribute__((ext_vector_type(8)));

#define C_IN   256
#define C_OUT  256
#define HW     4096   // 64*64
#define WI     64
#define HI     64
#define KK     9
#define KDIM   (KK * C_IN)     // 2304
#define NTILE  64              // spatial pixels per workgroup
#define KCH    32              // K-chunk (bf16 WMMA depth)
#define NCHUNK (KDIM / KCH)    // 72

__device__ __forceinline__ unsigned short f2bf(float f) {
    union { float f; unsigned u; } v; v.f = f;
    unsigned u = v.u + 0x7FFFu + ((v.u >> 16) & 1u);   // round-to-nearest-even
    return (unsigned short)(u >> 16);
}

// gfx1250 async bulk copy: global -> LDS, 16B per lane, tracked by ASYNCcnt.
__device__ __forceinline__ void async_load_b128(unsigned lds_byte_addr,
                                                unsigned long long gaddr) {
    asm volatile("global_load_async_to_lds_b128 %0, %1, off"
                 :: "v"(lds_byte_addr), "v"(gaddr) : "memory");
}
__device__ __forceinline__ void wait_asynccnt0() {
    asm volatile("s_wait_asynccnt 0" ::: "memory");
}

// ---------------------------------------------------------------------------
// Stage 1: offsets = conv3x3(x, offset_w) + offset_b   -> ws (f32)
// layout: offs[b][ch][p], ch = kk*2 + {0:y,1:x}
// ---------------------------------------------------------------------------
__global__ void offset_conv_kernel(const float* __restrict__ x,
                                   const float* __restrict__ ow,
                                   const float* __restrict__ ob,
                                   float* __restrict__ offs) {
    int id = blockIdx.x * blockDim.x + threadIdx.x;       // [0, 4*18*4096)
    if (id >= 4 * 18 * HW) return;
    int p  = id & (HW - 1);
    int ch = (id >> 12) % 18;
    int b  = id / (18 * HW);
    int oh = p >> 6, owi = p & 63;

    float acc = ob[ch];
    const float* xb = x + ((size_t)b * C_IN) * HW;
    const float* wb = ow + (size_t)ch * C_IN * KK;
    for (int c = 0; c < C_IN; ++c) {
        const float* plane = xb + (size_t)c * HW;
        const float* wk = wb + c * KK;
#pragma unroll
        for (int ky = 0; ky < 3; ++ky) {
#pragma unroll
            for (int kx = 0; kx < 3; ++kx) {
                int yy = oh + ky - 1, xx = owi + kx - 1;
                if (yy >= 0 && yy < HI && xx >= 0 && xx < WI)
                    acc += plane[yy * WI + xx] * wk[ky * 3 + kx];
            }
        }
    }
    offs[id] = acc;
}

// ---------------------------------------------------------------------------
// Stage 2: wbf[kk*65536 + o*256 + c] = bf16( weight[o*2304 + c*9 + kk] )
// ---------------------------------------------------------------------------
__global__ void weight_to_bf16_kernel(const float* __restrict__ w,
                                      unsigned short* __restrict__ wbf) {
    int id = blockIdx.x * blockDim.x + threadIdx.x;       // [0, 9*256*256)
    if (id >= KK * C_OUT * C_IN) return;
    int kk = id >> 16;
    int o  = (id >> 8) & 255;
    int c  = id & 255;
    wbf[id] = f2bf(w[(size_t)o * KDIM + c * KK + kk]);
}

// ---------------------------------------------------------------------------
// Stage 3: fused bilinear-im2col + GEMM via WMMA, double-buffered.
// grid: (64 pixel tiles, 4 batches), block: 256 threads (8 wave32).
// Workgroup computes out[b, 0..255, p0..p0+63].
// ---------------------------------------------------------------------------
__global__ void __launch_bounds__(256)
deform_gemm_kernel(const float* __restrict__ x,
                   const unsigned short* __restrict__ wbf,
                   const float* __restrict__ bias,
                   const float* __restrict__ offs,
                   float* __restrict__ out) {
    __shared__ __align__(64) unsigned short Aw[2][C_OUT][KCH];   // 2x16 KB weights
    __shared__ __align__(64) unsigned short Bt[2][NTILE][KCH];   // 2x 4 KB sampled^T
    __shared__ int4   cidx[KK][NTILE];                           // bilinear indices
    __shared__ float4 cwgt[KK][NTILE];                           // bilinear weights

    const int t    = threadIdx.x;
    const int lane = t & 31;
    const int wv   = t >> 5;                 // 0..7
    const int b    = blockIdx.y;
    const int p0   = blockIdx.x * NTILE;

    // --- precompute bilinear taps for this pixel tile -----------------------
    for (int i = t; i < KK * NTILE; i += 256) {
        int kk = i / NTILE, n = i % NTILE;
        int p = p0 + n, oh = p >> 6, owi = p & 63;
        int ky = kk / 3, kx = kk % 3;
        float off_y = offs[((size_t)b * 18 + kk * 2 + 0) * HW + p];
        float off_x = offs[((size_t)b * 18 + kk * 2 + 1) * HW + p];
        float py = (float)(oh + ky - 1) + off_y;
        float px = (float)(owi + kx - 1) + off_x;
        float y0 = floorf(py), x0 = floorf(px);
        float fy = py - y0, fx = px - x0;
        int y0i = (int)y0, x0i = (int)x0;
        int4 idx; float4 w;
#pragma unroll
        for (int cr = 0; cr < 4; ++cr) {
            int dy = cr >> 1, dx = cr & 1;
            int yy = y0i + dy, xx = x0i + dx;
            bool valid = (yy >= 0) & (yy < HI) & (xx >= 0) & (xx < WI);
            float wt = (dy ? fy : 1.f - fy) * (dx ? fx : 1.f - fx);
            int yc = min(max(yy, 0), HI - 1), xc = min(max(xx, 0), WI - 1);
            int id1 = yc * WI + xc;
            float wv2 = valid ? wt : 0.f;
            if (cr == 0) { idx.x = id1; w.x = wv2; }
            else if (cr == 1) { idx.y = id1; w.y = wv2; }
            else if (cr == 2) { idx.z = id1; w.z = wv2; }
            else { idx.w = id1; w.w = wv2; }
        }
        cidx[kk][n] = idx;
        cwgt[kk][n] = w;
    }
    __syncthreads();

    const float* xb = x + ((size_t)b * C_IN) * HW;

    // stage chunk `ch` (kk = ch/8, c0 = (ch%8)*32) into buffer `buf`
    auto stage = [&](int ch, int buf) {
        const int kk = ch >> 3;
        const int c0 = (ch & 7) * KCH;
        // A tile: async global->LDS, 4x b128 per thread (row t, 64B)
        {
            const unsigned short* gsrc = wbf + ((size_t)kk << 16) + t * C_IN + c0;
            unsigned ldsb = (unsigned)(unsigned long long)(uintptr_t)&Aw[buf][t][0];
#pragma unroll
            for (int q = 0; q < 4; ++q)
                async_load_b128(ldsb + q * 16,
                                (unsigned long long)(uintptr_t)(gsrc + q * 8));
        }
        // B^T tile: bilinear sample channel c0+j at tap (kk, pixel n)
#pragma unroll
        for (int r = 0; r < 8; ++r) {
            int e = t + 256 * r;                 // 0..2047
            int n = e >> 5, j = e & 31;
            const float* plane = xb + (size_t)(c0 + j) * HW;
            int4   id4 = cidx[kk][n];
            float4 w4  = cwgt[kk][n];
            float v = w4.x * plane[id4.x] + w4.y * plane[id4.y] +
                      w4.z * plane[id4.z] + w4.w * plane[id4.w];
            Bt[buf][n][j] = f2bf(v);
        }
    };

    // accumulators: wave wv owns rows [32*wv, 32*wv+32), 2(M) x 4(N) subtiles
    v8f acc[2][4];
#pragma unroll
    for (int mi = 0; mi < 2; ++mi)
#pragma unroll
        for (int ni = 0; ni < 4; ++ni)
            acc[mi][ni] = (v8f){0.f, 0.f, 0.f, 0.f, 0.f, 0.f, 0.f, 0.f};

    // prologue
    stage(0, 0);
    wait_asynccnt0();
    __syncthreads();

    const int klA = (lane < 16) ? 0 : 16;   // A K-base per half-wave
    const int klB = (lane < 16) ? 0 : 16;   // B K-base per half-wave

    for (int ch = 0; ch < NCHUNK; ++ch) {
        const int cur = ch & 1;
        const int nxt = cur ^ 1;
        if (ch + 1 < NCHUNK) stage(ch + 1, nxt);   // overlap with WMMAs below

#pragma unroll
        for (int mi = 0; mi < 2; ++mi) {
            int m = wv * 32 + mi * 16 + (lane & 15);
            v8bf alo = *(const v8bf*)&Aw[cur][m][klA];        // K = base+0..7
            v8bf ahi = *(const v8bf*)&Aw[cur][m][klA + 8];    // K = base+8..15
            v16bf a = __builtin_shufflevector(alo, ahi,
                      0, 1, 2, 3, 4, 5, 6, 7, 8, 9, 10, 11, 12, 13, 14, 15);
#pragma unroll
            for (int ni = 0; ni < 4; ++ni) {
                int n = ni * 16 + (lane & 15);
                v16bf bm = *(const v16bf*)&Bt[cur][n][klB];   // K contiguous
                acc[mi][ni] = __builtin_amdgcn_wmma_f32_16x16x32_bf16(
                    false, a, false, bm, (short)0, acc[mi][ni], false, false);
            }
        }

        wait_asynccnt0();    // next A tile resident in LDS
        __syncthreads();     // everyone done reading cur / writing nxt
    }

    // --- epilogue: D layout => VGPR r: M = r (lanes 0-15) / r+8 (lanes 16-31)
#pragma unroll
    for (int mi = 0; mi < 2; ++mi) {
#pragma unroll
        for (int ni = 0; ni < 4; ++ni) {
            int m_base = wv * 32 + mi * 16;
            int col = p0 + ni * 16 + (lane & 15);
#pragma unroll
            for (int r = 0; r < 8; ++r) {
                int row = m_base + ((lane < 16) ? r : r + 8);
                out[((size_t)b * C_OUT + row) * HW + col] = acc[mi][ni][r] + bias[row];
            }
        }
    }
}

// ---------------------------------------------------------------------------
extern "C" void kernel_launch(void* const* d_in, const int* in_sizes, int n_in,
                              void* d_out, int out_size, void* d_ws, size_t ws_size,
                              hipStream_t stream) {
    const float* x        = (const float*)d_in[0];
    const float* weight   = (const float*)d_in[1];
    const float* bias     = (const float*)d_in[2];
    const float* offset_w = (const float*)d_in[3];
    const float* offset_b = (const float*)d_in[4];
    float* out = (float*)d_out;

    float*          offs = (float*)d_ws;                                   // 1.18 MB
    unsigned short* wbf  = (unsigned short*)((char*)d_ws + (2u << 20));    // 1.18 MB

    // Stage 1: offset conv
    {
        int total = 4 * 18 * HW;
        offset_conv_kernel<<<(total + 255) / 256, 256, 0, stream>>>(
            x, offset_w, offset_b, offs);
    }
    // Stage 2: weight permute + bf16
    {
        int total = KK * C_OUT * C_IN;
        weight_to_bf16_kernel<<<(total + 255) / 256, 256, 0, stream>>>(weight, wbf);
    }
    // Stage 3: fused deformable GEMM (WMMA, async-LDS staging)
    {
        dim3 grid(HW / NTILE, 4);
        deform_gemm_kernel<<<grid, 256, 0, stream>>>(x, wbf, bias, offs, out);
    }
}
